// ContinuousPlanarFlow_41618233098888
// MI455X (gfx1250) — matrix operand order repaired and checked by reading, not compiled, not run on hardware
//
#include <hip/hip_runtime.h>
#include <stdint.h>

#define Bn 65536
#define Dn 256
#define Wn 1024

// padded LDS row strides (in 16-bit elements) to avoid bank conflicts:
// 264 elems = 528 B = 132 dwords -> lane rotation 4 banks (rows 256 wide)
// 40  elems =  80 B =  20 dwords -> lane rotation 20 banks (rows 32 wide)
#define ZP 264
#define WP 40

typedef __bf16 v16bf __attribute__((ext_vector_type(16)));
typedef float  v8f   __attribute__((ext_vector_type(8)));
typedef unsigned int u32x4 __attribute__((ext_vector_type(4)));
typedef float  f32x4 __attribute__((ext_vector_type(4)));

union FragU { u32x4 q[2]; v16bf v; };

__device__ __forceinline__ unsigned short f2bf(float x) {
    union { float f; unsigned int u; } c; c.f = x;
    unsigned int r = c.u + 0x7FFFu + ((c.u >> 16) & 1u);   // round-to-nearest-even
    return (unsigned short)(r >> 16);
}

// ---- prep: fp32 -> bf16 copies of W1, W2 (row-major, L2-resident) ----
__global__ void prep_convert(const float* __restrict__ W1,
                             const float* __restrict__ W2,
                             unsigned short* __restrict__ w1b,
                             unsigned short* __restrict__ w2b) {
    int i = blockIdx.x * blockDim.x + threadIdx.x;
    if (i < Wn * Dn) {
        w1b[i] = f2bf(W1[i]);
        w2b[i] = f2bf(W2[i]);
    }
}

// ---- prep: M[j] = sum_i W1[j,i] * W2[i,j] ----
__global__ void prep_M(const float* __restrict__ W1,
                       const float* __restrict__ W2,
                       float* __restrict__ M) {
    __shared__ float red[256];
    int j = blockIdx.x;
    int t = threadIdx.x;
    red[t] = W1[j * Dn + t] * W2[t * Wn + j];
    __syncthreads();
    for (int s = 128; s > 0; s >>= 1) {
        if (t < s) red[t] += red[t + s];
        __syncthreads();
    }
    if (t == 0) M[j] = red[0];
}

// issue async copies of one 32-hidden weight chunk into LDS (whole block: 2048 b128s)
__device__ __forceinline__ void issue_chunk(const unsigned short* __restrict__ w1b,
                                            const unsigned short* __restrict__ w2b,
                                            unsigned short* w1d, unsigned short* w2d,
                                            int j0, int tid) {
    // W1 rows j0..j0+31, 256 elems each: 1024 x 16B units
    #pragma unroll
    for (int u = 0; u < 8; ++u) {
        int v = tid + u * 128;
        int r = v >> 5, p = v & 31;
        const unsigned short* g = w1b + (size_t)(j0 + r) * Dn + p * 8;
        unsigned lds = (unsigned)(uintptr_t)(w1d + r * ZP + p * 8);
        asm volatile("global_load_async_to_lds_b128 %0, %1, off"
                     :: "v"(lds), "v"((unsigned long long)(uintptr_t)g) : "memory");
    }
    // W2 rows i=0..255, 32 elems at col j0: 1024 x 16B units
    #pragma unroll
    for (int u = 0; u < 8; ++u) {
        int v = tid + u * 128;
        int i = v >> 2, p = v & 3;
        const unsigned short* g = w2b + (size_t)i * Wn + j0 + p * 8;
        unsigned lds = (unsigned)(uintptr_t)(w2d + i * WP + p * 8);
        asm volatile("global_load_async_to_lds_b128 %0, %1, off"
                     :: "v"(lds), "v"((unsigned long long)(uintptr_t)g) : "memory");
    }
}

// ---- fused: pre = zW1^T+b1; h=softplus; dz=hW2^T; trace=sigmoid(pre).M ----
__launch_bounds__(128)
__global__ void flow_fused(const float* __restrict__ z,
                           const float* __restrict__ b1,
                           const unsigned short* __restrict__ w1b,
                           const unsigned short* __restrict__ w2b,
                           const float* __restrict__ Mv,
                           float* __restrict__ out) {
    __shared__ unsigned short zt[4][16][ZP];      // per-wave z tile, bf16 (padded)
    __shared__ unsigned short ht[4][16][WP];      // per-wave h tile, bf16 (padded)
    __shared__ unsigned short w1s[2][32 * ZP];    // double-buffered W1 chunk
    __shared__ unsigned short w2s[2][256 * WP];   // double-buffered W2 chunk

    const int tid = threadIdx.x;
    const int wid = tid >> 5;
    const int lane = tid & 31;
    const int l16 = lane & 15;
    const int hi  = lane >> 4;                    // lane half: 0 or 1
    const int row_base = blockIdx.x * 64 + wid * 16;

    // prologue: kick off weight chunk 0 while we stage z
    issue_chunk(w1b, w2b, w1s[0], w2s[0], 0, tid);

    // stage this wave's 16x256 z tile into LDS as bf16
    {
        const float* zp = z + (size_t)row_base * Dn;
        for (int t = lane * 4; t < 16 * 256; t += 32 * 4) {
            f32x4 v = *(const f32x4*)(zp + t);
            int r = t >> 8, c = t & 255;
            unsigned int p0 = (unsigned int)f2bf(v.x) | ((unsigned int)f2bf(v.y) << 16);
            unsigned int p1 = (unsigned int)f2bf(v.z) | ((unsigned int)f2bf(v.w) << 16);
            *(unsigned int*)&zt[wid][r][c + 0] = p0;
            *(unsigned int*)&zt[wid][r][c + 2] = p1;
        }
    }
    __syncthreads();

    v8f dz[16];
    float trv[8];
    #pragma unroll
    for (int n = 0; n < 16; ++n)
        #pragma unroll
        for (int v = 0; v < 8; ++v) dz[n][v] = 0.f;
    #pragma unroll
    for (int v = 0; v < 8; ++v) trv[v] = 0.f;

    // A-fragment source: 16-bit A 16x32 layout -> two contiguous 8-elem runs
    const unsigned short* ztp = &zt[wid][l16][hi * 8];

    for (int jc = 0; jc < Wn / 32; ++jc) {
        const int j0 = jc * 32;

        // double-buffer: issue next chunk, then wait for current one
        if (jc < Wn / 32 - 1) {
            issue_chunk(w1b, w2b, w1s[(jc + 1) & 1], w2s[(jc + 1) & 1],
                        j0 + 32, tid);
            asm volatile("s_wait_asynccnt 16" ::: "memory");
        } else {
            asm volatile("s_wait_asynccnt 0" ::: "memory");
        }
        __syncthreads();   // all waves' portions of current chunk are in LDS

        const unsigned short* w1c = w1s[jc & 1];
        const unsigned short* w2c = w2s[jc & 1];

        v8f pre0, pre1;
        #pragma unroll
        for (int v = 0; v < 8; ++v) { pre0[v] = 0.f; pre1[v] = 0.f; }

        // B1 fragments from LDS: column n = hidden j; two 8-elem K runs (table order)
        const unsigned short* w1r0 = w1c + l16 * ZP + hi * 8;
        const unsigned short* w1r1 = w1r0 + 16 * ZP;

        #pragma unroll
        for (int kc = 0; kc < 8; ++kc) {
            FragU a, bb0, bb1;
            a.q[0]   = *(const u32x4*)(ztp  + kc * 32);
            a.q[1]   = *(const u32x4*)(ztp  + kc * 32 + 16);
            bb0.q[0] = *(const u32x4*)(w1r0 + kc * 32);
            bb0.q[1] = *(const u32x4*)(w1r0 + kc * 32 + 16);
            bb1.q[0] = *(const u32x4*)(w1r1 + kc * 32);
            bb1.q[1] = *(const u32x4*)(w1r1 + kc * 32 + 16);
            pre0 = __builtin_amdgcn_wmma_f32_16x16x32_bf16(
                false, a.v, false, bb0.v, (short)0, pre0, false, false);
            pre1 = __builtin_amdgcn_wmma_f32_16x16x32_bf16(
                false, a.v, false, bb1.v, (short)0, pre1, false, false);
        }

        // bias + softplus + sigmoid.M ; write h (bf16) to LDS for transpose
        const float bv0 = b1[j0 + l16];
        const float bv1 = b1[j0 + 16 + l16];
        const float m0  = Mv[j0 + l16];
        const float m1  = Mv[j0 + 16 + l16];
        #pragma unroll
        for (int v = 0; v < 8; ++v) {
            float p0 = pre0[v] + bv0;
            float p1 = pre1[v] + bv1;
            float e0 = __expf(p0), e1 = __expf(p1);
            float h0 = (p0 > 15.f) ? p0 : __logf(1.f + e0);
            float h1 = (p1 > 15.f) ? p1 : __logf(1.f + e1);
            float s0 = (p0 > 15.f) ? 1.f : e0 / (1.f + e0);
            float s1 = (p1 > 15.f) ? 1.f : e1 / (1.f + e1);
            trv[v] += s0 * m0 + s1 * m1;
            int m = v + hi * 8;
            ht[wid][m][l16]      = f2bf(h0);
            ht[wid][m][16 + l16] = f2bf(h1);
        }
        asm volatile("s_wait_dscnt 0" ::: "memory");  // C-layout -> A-layout via LDS

        FragU ah;
        ah.q[0] = *(const u32x4*)&ht[wid][l16][hi * 8];
        ah.q[1] = *(const u32x4*)&ht[wid][l16][hi * 8 + 16];

        // GEMM2: dz[16x256] += h(16x32) * W2^T(32x256), B2 from LDS
        #pragma unroll
        for (int nc = 0; nc < 16; ++nc) {
            FragU b2;
            const unsigned short* p = w2c + (nc * 16 + l16) * WP + hi * 8;
            b2.q[0] = *(const u32x4*)(p);
            b2.q[1] = *(const u32x4*)(p + 16);
            dz[nc] = __builtin_amdgcn_wmma_f32_16x16x32_bf16(
                false, ah.v, false, b2.v, (short)0, dz[nc], false, false);
        }
        __syncthreads();   // done reading this buffer before it is refilled
    }

    // trace: butterfly reduce across each 16-lane group
    #pragma unroll
    for (int v = 0; v < 8; ++v) {
        float t = trv[v];
        t += __shfl_xor(t, 1, 32);
        t += __shfl_xor(t, 2, 32);
        t += __shfl_xor(t, 4, 32);
        t += __shfl_xor(t, 8, 32);
        trv[v] = t;
    }
    if (l16 == 0) {
        float* tout = out + (size_t)Bn * Dn + row_base + hi * 8;
        #pragma unroll
        for (int v = 0; v < 8; ++v) tout[v] = -trv[v];
    }

    // dz output: element (m, n) -> row m in VGPR index, col n in lane
    float* op = out + (size_t)row_base * Dn;
    #pragma unroll
    for (int nc = 0; nc < 16; ++nc) {
        #pragma unroll
        for (int v = 0; v < 8; ++v) {
            op[(size_t)(v + hi * 8) * Dn + nc * 16 + l16] = dz[nc][v];
        }
    }
}

extern "C" void kernel_launch(void* const* d_in, const int* in_sizes, int n_in,
                              void* d_out, int out_size, void* d_ws, size_t ws_size,
                              hipStream_t stream) {
    // inputs: t(unused), z[B,D], W1[W,D], b1[W], W2[D,W]
    const float* z  = (const float*)d_in[1];
    const float* W1 = (const float*)d_in[2];
    const float* b1 = (const float*)d_in[3];
    const float* W2 = (const float*)d_in[4];
    float* out = (float*)d_out;

    unsigned short* w1b = (unsigned short*)d_ws;                 // 512 KB
    unsigned short* w2b = w1b + (size_t)Wn * Dn;                 // 512 KB
    float*          Mv  = (float*)(w2b + (size_t)Dn * Wn);       // 4 KB

    prep_convert<<<(Wn * Dn + 255) / 256, 256, 0, stream>>>(W1, W2, w1b, w2b);
    prep_M<<<Wn, 256, 0, stream>>>(W1, W2, Mv);
    flow_fused<<<Bn / 64, 128, 0, stream>>>(z, b1, w1b, w2b, Mv, out);
}
